// Repeat_Recommendation_Decoder_69861938037631
// MI455X (gfx1250) — matrix operand order repaired and checked by reading, not compiled, not run on hardware
//
#include <hip/hip_runtime.h>
#include <hip/hip_bf16.h>
#include <math.h>

typedef __attribute__((ext_vector_type(16))) _Float16 v16h;
typedef __attribute__((ext_vector_type(8)))  float    v8f;
typedef __attribute__((ext_vector_type(4)))  float    f4;

#define Bsz 1024
#define Ssz 50
#define Hsz 64
#define Vsz 100000
#define NEG_INF -1e9f

// ---------------------------------------------------------------------------
// D(R x 64) = X(R x 64) @ W^T(64 x 64), f16 WMMA with f32 accumulate.
// SCORES=false: store D raw (used for L = last_memory @ Wr^T).
// SCORES=true : fused epilogue  score[r] = sum_n tanh(D[r,n] + L[b,n]) * Vr[n]
//               with b = r / S (rows are the flattened (B,S) dimension).
// Each wave (32 lanes) owns one 16-row M-tile; 8 waves per block; block stages
// W as f16 in LDS.
// ---------------------------------------------------------------------------
template <bool SCORES>
__global__ void wmma_gemm64_kernel(const float* __restrict__ X,
                                   const float* __restrict__ W,
                                   const float* __restrict__ Lbuf,
                                   const float* __restrict__ Vr,
                                   float* __restrict__ outD,
                                   int R) {
  __shared__ _Float16 Wl[64 * 64];
  __shared__ float    Vl[64];

  const int tid = threadIdx.x;          // 0..255
  // Stage W (row n, col h) as f16 in LDS.
  #pragma unroll
  for (int i = tid; i < 64 * 64; i += 256) Wl[i] = (_Float16)W[i];
  if (SCORES && tid < 64) Vl[tid] = Vr[tid];
  __syncthreads();

  const int wave   = tid >> 5;
  const int lane   = tid & 31;
  const int half16 = lane >> 4;         // 0: K-lo half, 1: K-hi half (A layout)
  const int l16    = lane & 15;
  const long long tile = (long long)blockIdx.x * 8 + wave;
  const long long r0   = tile * 16;
  if (r0 >= R) return;

  // ---- A fragments (16x32 f16 layout, ISA 7.12.2): lane holds row M=l16;
  // lanes 0-15 cover K in [0..7]u[16..23], lanes 16-31 cover [8..15]u[24..31].
  const float* rowp = X + (r0 + l16) * 64;
  const int kb = half16 * 8;
  v16h a0, a1;
  #pragma unroll
  for (int j = 0; j < 8; ++j) {
    a0[j]     = (_Float16)rowp[kb + j];
    a0[8 + j] = (_Float16)rowp[16 + kb + j];
    a1[j]     = (_Float16)rowp[32 + kb + j];
    a1[8 + j] = (_Float16)rowp[48 + kb + j];
  }

  float score[8];
  if (SCORES) {
    #pragma unroll
    for (int v = 0; v < 8; ++v) score[v] = 0.f;
  }

  // ---- 4 N-tiles of 16 columns each.
  #pragma unroll
  for (int nt = 0; nt < 4; ++nt) {
    // B fragment (32x16 f16): lane holds column n = nt*16 + l16; lanes 0-15
    // cover K in [0..7]u[16..23], lanes 16-31 cover [8..15]u[24..31] (mirror of A layout).
    // B[k][n] = W[n][k] (we need X @ W^T).
    const _Float16* wr = &Wl[(nt * 16 + l16) * 64 + half16 * 8];
    v16h b0, b1;
    #pragma unroll
    for (int j = 0; j < 8; ++j) { b0[j] = wr[j]; b0[8 + j] = wr[16 + j]; b1[j] = wr[32 + j]; b1[8 + j] = wr[48 + j]; }

    v8f c = {};
    c = __builtin_amdgcn_wmma_f32_16x16x32_f16(false, a0, false, b0,
                                               (short)0, c, false, false);
    c = __builtin_amdgcn_wmma_f32_16x16x32_f16(false, a1, false, b1,
                                               (short)0, c, false, false);

    // C/D layout: lane n=l16 holds M = v + half16*8 in VGPR v.
    const int n = nt * 16 + l16;
    if (SCORES) {
      const float vr = Vl[n];
      #pragma unroll
      for (int v = 0; v < 8; ++v) {
        const long long gr = r0 + v + half16 * 8;   // flattened (b,s) row
        const long long b  = gr / Ssz;
        const float t = tanhf(c[v] + Lbuf[b * 64 + n]);
        score[v] += t * vr;
      }
    } else {
      #pragma unroll
      for (int v = 0; v < 8; ++v)
        outD[(r0 + v + half16 * 8) * 64 + n] = c[v];
    }
  }

  if (SCORES) {
    // Reduce over the 16 lanes of each half (full sum over n=0..63 for each
    // of this half's 8 rows). XOR masks 1..8 keep halves separate (wave32).
    #pragma unroll
    for (int m = 1; m <= 8; m <<= 1)
      #pragma unroll
      for (int v = 0; v < 8; ++v)
        score[v] += __shfl_xor(score[v], m, 32);
    if (l16 == 0) {
      #pragma unroll
      for (int v = 0; v < 8; ++v)
        outD[r0 + half16 * 8 + v] = score[v];
    }
  }
}

// ---------------------------------------------------------------------------
// Masked softmax over S=50 per batch row. One wave per row, 4 rows per block.
// Vr_b is a uniform shift -> cancels in softmax, omitted.
// ---------------------------------------------------------------------------
__global__ void softmax_kernel(const float* __restrict__ scores,
                               const unsigned char* __restrict__ mask,
                               float* __restrict__ att) {
  const int wave = threadIdx.x >> 5;
  const int lane = threadIdx.x & 31;
  const int b = blockIdx.x * 4 + wave;
  if (b >= Bsz) return;

  const int i0 = lane, i1 = lane + 32;
  float x0 = -3.0e38f, x1 = -3.0e38f;
  if (i0 < Ssz) x0 = mask[b * Ssz + i0] ? NEG_INF : scores[b * Ssz + i0];
  if (i1 < Ssz) x1 = mask[b * Ssz + i1] ? NEG_INF : scores[b * Ssz + i1];

  float m = fmaxf(x0, x1);
  #pragma unroll
  for (int d = 16; d >= 1; d >>= 1) m = fmaxf(m, __shfl_xor(m, d, 32));

  const float e0 = (i0 < Ssz) ? __expf(x0 - m) : 0.f;
  const float e1 = (i1 < Ssz) ? __expf(x1 - m) : 0.f;
  float s = e0 + e1;
  #pragma unroll
  for (int d = 16; d >= 1; d >>= 1) s += __shfl_xor(s, d, 32);
  const float inv = __frcp_rn(s);

  if (i0 < Ssz) att[b * Ssz + i0] = e0 * inv;
  if (i1 < Ssz) att[b * Ssz + i1] = e1 * inv;
}

// ---------------------------------------------------------------------------
// Zero the 409.6 MB output with non-temporal b128 streaming stores (output is
// > 2x the 192 MB L2; no reuse worth caching before the sparse scatter).
// This kernel is the roofline bottleneck (~18 us at 23.3 TB/s).
// ---------------------------------------------------------------------------
__global__ void zero_kernel(f4* __restrict__ out, long long n4) {
  long long i = (long long)blockIdx.x * blockDim.x + threadIdx.x;
  const long long stride = (long long)gridDim.x * blockDim.x;
  const f4 z = {0.f, 0.f, 0.f, 0.f};
  for (; i < n4; i += stride) __builtin_nontemporal_store(z, &out[i]);
}

// ---------------------------------------------------------------------------
// Scatter: out[b, seq_item[b,s]] += att[b,s]. Duplicates within a row are
// possible -> atomic f32 add (51200 atomics total, negligible).
// ---------------------------------------------------------------------------
__global__ void scatter_kernel(const float* __restrict__ att,
                               const long long* __restrict__ seq,
                               float* __restrict__ out) {
  const int i = blockIdx.x * blockDim.x + threadIdx.x;
  if (i >= Bsz * Ssz) return;
  const long long b = i / Ssz;
  const long long item = seq[i];
  atomicAdd(out + b * (long long)Vsz + item, att[i]);
}

extern "C" void kernel_launch(void* const* d_in, const int* in_sizes, int n_in,
                              void* d_out, int out_size, void* d_ws, size_t ws_size,
                              hipStream_t stream) {
  const float*          all_memory  = (const float*)d_in[0];          // (B,S,H)
  const float*          last_memory = (const float*)d_in[1];          // (B,H)
  const long long*      seq_item    = (const long long*)d_in[2];      // (B,S) i64
  const unsigned char*  mask        = (const unsigned char*)d_in[3];  // (B,S) bool
  const float*          Ur_w        = (const float*)d_in[4];          // (H,H)
  const float*          Wr_w        = (const float*)d_in[5];          // (H,H)
  const float*          Vr_w        = (const float*)d_in[6];          // (H,)
  // d_in[7] = Vr_b: uniform pre-softmax shift, cancels in softmax.

  float* Lbuf   = (float*)d_ws;                       // B*64 floats
  float* scores = Lbuf + (size_t)Bsz * 64;            // B*S floats
  float* att    = scores + (size_t)Bsz * Ssz;         // B*S floats
  float* out    = (float*)d_out;                      // B*V floats

  // 1) L = last_memory @ Wr^T : 1024 rows -> 64 tiles -> 8 blocks x 8 waves.
  wmma_gemm64_kernel<false><<<dim3(8), dim3(256), 0, stream>>>(
      last_memory, Wr_w, nullptr, nullptr, Lbuf, Bsz);

  // 2) scores = Vr . tanh(all_memory @ Ur^T + L) : 51200 rows -> 3200 tiles.
  wmma_gemm64_kernel<true><<<dim3(400), dim3(256), 0, stream>>>(
      all_memory, Ur_w, Lbuf, Vr_w, scores, Bsz * Ssz);

  // 3) masked softmax over S per batch row.
  softmax_kernel<<<dim3(Bsz / 4), dim3(128), 0, stream>>>(scores, mask, att);

  // 4) zero the (B,V) output (dominant cost, pure store bandwidth).
  const long long n4 = ((long long)Bsz * Vsz) / 4;    // 25.6M float4
  zero_kernel<<<dim3(12800), dim3(256), 0, stream>>>((f4*)out, n4);

  // 5) scatter-add attention weights by item id.
  scatter_kernel<<<dim3((Bsz * Ssz + 255) / 256), dim3(256), 0, stream>>>(
      att, seq_item, out);
}
